// PMBNNAlwaysPlastic_5196910428905
// MI455X (gfx1250) — compile-verified, matrix operand order checked by reading, not run on hardware
//
#include <hip/hip_runtime.h>

typedef __attribute__((ext_vector_type(16))) _Float16 v16h;
typedef __attribute__((ext_vector_type(8)))  _Float16 v8h;
typedef __attribute__((ext_vector_type(4)))  _Float16 v4h;
typedef __attribute__((ext_vector_type(8)))  float    v8f;

#define NROWS    8192
#define D_LAT    8
#define N_CENT   48
#define CHANNELS 64
#define HID      256
#define IN_DIM   784
#define LPITCH   40   // LDS pitch in halves: 32 data + 8 pad (breaks 4-way bank conflict)

// K-index striping for 16-bit WMMA A fragments (ISA 7.12.2, 16-bit A 16x32):
// lane group g = lane>>4; element e -> K = (e&7) + 8*g + 16*(e>>3)
__device__ __forceinline__ int wk_idx(int e, int g) {
    return (e & 7) + (g << 3) + ((e >> 3) << 4);
}

__device__ __forceinline__ v16h cat8(v8h lo, v8h hi) {
    return __builtin_shufflevector(lo, hi, 0,1,2,3,4,5,6,7,8,9,10,11,12,13,14,15);
}

// low 32 bits of a generic LDS pointer = workgroup-relative LDS byte offset
__device__ __forceinline__ unsigned lds_off(const void* p) {
    return (unsigned)(unsigned long long)(uintptr_t)p;
}

// async 16B copy global -> LDS, GVS form (SGPR64 base + per-lane u32 offset)
__device__ __forceinline__ void async_copy_b128(unsigned lds_byte, unsigned g_byte,
                                                unsigned long long base) {
    asm volatile("global_load_async_to_lds_b128 %0, %1, %2"
                 :: "v"(lds_byte), "v"(g_byte), "s"(base) : "memory");
}
// async 16B copy global -> LDS, GV form (per-lane 64-bit address; allows
// per-lane base selection so every wave issues a uniform async count)
__device__ __forceinline__ void async_copy_b128_gv(unsigned lds_byte,
                                                   unsigned long long gaddr) {
    asm volatile("global_load_async_to_lds_b128 %0, %1, off"
                 :: "v"(lds_byte), "v"(gaddr) : "memory");
}

// ---------------------------------------------------------------------------
// Encoder GEMM1: h1 = tanh(x @ W1 + b1)  [8192x784]x[784x256] -> f16 [8192x256]
// WG = 4 waves = 16 rows x 64 cols; 24 guard-free K-chunks + 16-wide tail.
// B tile stored transposed in LDS so both fragment gathers are ds_load_b128.
// ---------------------------------------------------------------------------
__global__ void __launch_bounds__(128)
enc1_kernel(const float* __restrict__ x, const float* __restrict__ W1,
            const float* __restrict__ b1, _Float16* __restrict__ h1) {
    __shared__ __align__(16) _Float16 At[16][LPITCH];
    __shared__ __align__(16) _Float16 BtT[64][LPITCH];   // [col][k]
    const int tid  = threadIdx.x;
    const int wave = tid >> 5, lane = tid & 31;
    const int g = lane >> 4, ln = lane & 15;
    const int rowBase = blockIdx.x * 16, colBase = blockIdx.y * 64;
    const int ar = tid >> 3, ac = (tid & 7) * 4;     // A staging: row, col4

    v8f acc = {};

    for (int kc = 0; kc < 768; kc += 32) {           // guard-free main chunks
        __syncthreads();
        {
            const float4 xa = *(const float4*)(x + (rowBase + ar) * IN_DIM + kc + ac);
            v4h p = { (_Float16)xa.x, (_Float16)xa.y, (_Float16)xa.z, (_Float16)xa.w };
            *(v4h*)&At[ar][ac] = p;
        }
        #pragma unroll
        for (int i = 0; i < 4; ++i) {
            int grp = tid + i * 128;                 // 512 float4 groups (32k x 16c4)
            int r = grp >> 4, c0 = (grp & 15) * 4;
            const float4 wv = *(const float4*)(W1 + (kc + r) * HID + colBase + c0);
            BtT[c0 + 0][r] = (_Float16)wv.x;
            BtT[c0 + 1][r] = (_Float16)wv.y;
            BtT[c0 + 2][r] = (_Float16)wv.z;
            BtT[c0 + 3][r] = (_Float16)wv.w;
        }
        if (kc + 32 < 768) {                         // prefetch next chunk
            __builtin_prefetch(x + (rowBase + ar) * IN_DIM + kc + 32 + ac, 0, 1);
            __builtin_prefetch(W1 + (kc + 32 + (tid >> 4)) * HID + colBase, 0, 1);
        }
        __syncthreads();
        v8h a01 = *(const v8h*)&At[ln][8 * g];
        v8h a23 = *(const v8h*)&At[ln][16 + 8 * g];
        v8h b01 = *(const v8h*)&BtT[wave * 16 + ln][8 * g];
        v8h b23 = *(const v8h*)&BtT[wave * 16 + ln][16 + 8 * g];
        acc = __builtin_amdgcn_wmma_f32_16x16x32_f16(false, cat8(a01, a23),
                                                     false, cat8(b01, b23),
                                                     (short)0, acc, false, false);
    }
    {   // tail chunk kc = 768, 16 valid k
        const int kc = 768;
        __syncthreads();
        {
            v4h p = {};
            if (ac < 16) {
                const float4 xa = *(const float4*)(x + (rowBase + ar) * IN_DIM + kc + ac);
                p = (v4h){ (_Float16)xa.x, (_Float16)xa.y, (_Float16)xa.z, (_Float16)xa.w };
            }
            *(v4h*)&At[ar][ac] = p;
        }
        #pragma unroll
        for (int i = 0; i < 4; ++i) {
            int grp = tid + i * 128;
            int r = grp >> 4, c0 = (grp & 15) * 4;
            float4 wv = {0.f, 0.f, 0.f, 0.f};
            if (r < 16) wv = *(const float4*)(W1 + (kc + r) * HID + colBase + c0);
            BtT[c0 + 0][r] = (_Float16)wv.x;
            BtT[c0 + 1][r] = (_Float16)wv.y;
            BtT[c0 + 2][r] = (_Float16)wv.z;
            BtT[c0 + 3][r] = (_Float16)wv.w;
        }
        __syncthreads();
        v8h a01 = *(const v8h*)&At[ln][8 * g];
        v8h a23 = *(const v8h*)&At[ln][16 + 8 * g];
        v8h b01 = *(const v8h*)&BtT[wave * 16 + ln][8 * g];
        v8h b23 = *(const v8h*)&BtT[wave * 16 + ln][16 + 8 * g];
        acc = __builtin_amdgcn_wmma_f32_16x16x32_f16(false, cat8(a01, a23),
                                                     false, cat8(b01, b23),
                                                     (short)0, acc, false, false);
    }
    const int col = colBase + wave * 16 + ln;
    #pragma unroll
    for (int r = 0; r < 8; ++r) {
        int row = rowBase + r + 8 * g;
        h1[row * HID + col] = (_Float16)tanhf(acc[r] + b1[col]);
    }
}

// ---------------------------------------------------------------------------
// Encoder GEMM2: z = h1 @ W2 + b2  (N=8, scalar; W2 staged in LDS, h1 read b128)
// ---------------------------------------------------------------------------
__global__ void __launch_bounds__(256)
enc2_kernel(const _Float16* __restrict__ h1, const float* __restrict__ W2,
            const float* __restrict__ b2, float* __restrict__ z) {
    __shared__ float W2s[HID * D_LAT];
    int tid = threadIdx.x;
    for (int i = tid; i < HID * D_LAT; i += 256) W2s[i] = W2[i];
    __syncthreads();
    int idx = blockIdx.x * 256 + tid;               // 8192*8
    int row = idx >> 3, d = idx & 7;
    const v8h* hp = (const v8h*)(h1 + row * HID);
    float s = b2[d];
    for (int kk = 0; kk < HID / 8; ++kk) {
        v8h hv = hp[kk];
        #pragma unroll
        for (int q = 0; q < 8; ++q)
            s += (float)hv[q] * W2s[(kk * 8 + q) * D_LAT + d];
    }
    z[idx] = s;
}

// ---------------------------------------------------------------------------
// PM field: 4 gradient-flow steps per call, one thread per row
// ---------------------------------------------------------------------------
__global__ void __launch_bounds__(256)
pm_kernel(float* __restrict__ z, const float* __restrict__ centers,
          const float* __restrict__ mus) {
    __shared__ float Cs[N_CENT][D_LAT];
    __shared__ float Ms[N_CENT];
    int tid = threadIdx.x;
    for (int i = tid; i < N_CENT * D_LAT; i += 256) Cs[i >> 3][i & 7] = centers[i];
    if (tid < N_CENT) Ms[tid] = mus[tid];
    __syncthreads();

    int row = blockIdx.x * 256 + tid;
    float zz[D_LAT];
    #pragma unroll
    for (int d = 0; d < D_LAT; ++d) zz[d] = z[row * D_LAT + d];

    for (int s = 0; s < 4; ++s) {
        float n = 1.f;
        float ga[D_LAT] = {0.f, 0.f, 0.f, 0.f, 0.f, 0.f, 0.f, 0.f};
        for (int c = 0; c < N_CENT; ++c) {
            float diff[D_LAT];
            float r2 = 1e-4f;
            #pragma unroll
            for (int d = 0; d < D_LAT; ++d) {
                diff[d] = zz[d] - Cs[c][d];
                r2 += diff[d] * diff[d];
            }
            float ir = rsqrtf(r2);          // 1/r
            float mu = Ms[c];
            n += mu * ir;
            float coef = mu * ir * ir * ir; // mu / (r * r2)
            #pragma unroll
            for (int d = 0; d < D_LAT; ++d) ga[d] -= coef * diff[d];
        }
        float sc = 0.12f / n;               // DT*BETA / n
        #pragma unroll
        for (int d = 0; d < D_LAT; ++d)
            zz[d] = fminf(fmaxf(zz[d] + sc * ga[d], -3.f), 3.f);
    }
    #pragma unroll
    for (int d = 0; d < D_LAT; ++d) z[row * D_LAT + d] = zz[d];
}

// ---------------------------------------------------------------------------
// Projection: h_mid = 0.9*h_prev + 0.1*tanh(z@projW + projb).
// Block covers 64 rows x 64 ch; writes row-major h_mid (f32) AND transposed
// h16T[ch][row] (f16) with contiguous 32B stores per thread, plus ||z||^2.
// ---------------------------------------------------------------------------
__global__ void __launch_bounds__(256)
proj_kernel(const float* __restrict__ z, const float* __restrict__ projW,
            const float* __restrict__ projb, const float* __restrict__ h_in,
            float* __restrict__ h_mid, _Float16* __restrict__ h16T,
            float* __restrict__ sq, int first) {
    __shared__ float Wp[D_LAT][CHANNELS];
    __shared__ float pb[CHANNELS];
    __shared__ float zs[64][D_LAT];
    int tid = threadIdx.x;
    for (int i = tid; i < D_LAT * CHANNELS; i += 256) Wp[i >> 6][i & 63] = projW[i];
    if (tid < CHANNELS) pb[tid] = projb[tid];
    int rowBase = blockIdx.x * 64;
    for (int i = tid; i < 64 * D_LAT; i += 256) zs[i >> 3][i & 7] = z[rowBase * D_LAT + i];
    __syncthreads();

    int ch = tid & 63, rg = tid >> 6;               // rg in 0..3 -> rows rg*16..+15
    v8h hh0 = {}, hh1 = {};
    #pragma unroll
    for (int q = 0; q < 16; ++q) {
        int lr = rg * 16 + q;
        int row = rowBase + lr;
        float a = pb[ch];
        #pragma unroll
        for (int d = 0; d < D_LAT; ++d) a += zs[lr][d] * Wp[d][ch];
        int idx = row * CHANNELS + ch;
        float hpv = first ? 0.f : h_in[idx];
        float hm = 0.9f * hpv + 0.1f * tanhf(a);
        h_mid[idx] = hm;
        if (q < 8) hh0[q] = (_Float16)hm; else hh1[q - 8] = (_Float16)hm;
    }
    *(v8h*)(h16T + ch * NROWS + rowBase + rg * 16)     = hh0;
    *(v8h*)(h16T + ch * NROWS + rowBase + rg * 16 + 8) = hh1;

    if (ch == 0) {
        #pragma unroll
        for (int q = 0; q < 16; ++q) {
            int lr = rg * 16 + q;
            float s = 0.f;
            #pragma unroll
            for (int d = 0; d < D_LAT; ++d) s += zs[lr][d] * zs[lr][d];
            sq[rowBase + lr] = s;
        }
    }
}

// ---------------------------------------------------------------------------
// Fused lateral E/I: h_out = h_mid + GAIN * (K_unnorm @ h_mid) / (rowsum + eps)
// Flash-style with DOUBLE-BUFFERED async global->LDS staging: batch n+1 is
// issued before computing on batch n; s_wait_asynccnt 3 retires only the
// oldest batch (async completion is in-order; every wave issues exactly 3
// async instructions per batch - lanes 72..127 copy into a dummy LDS sink).
// ---------------------------------------------------------------------------
__global__ void __launch_bounds__(128)
lateral_kernel(const float* __restrict__ z, const float* __restrict__ sq,
               const _Float16* __restrict__ h16T, const float* __restrict__ h_mid,
               float* __restrict__ h_out) {
    __shared__ __align__(16) float    zj[2][32][D_LAT];
    __shared__ __align__(16) float    sj[2][32];
    __shared__ __align__(16) _Float16 hjT[2][CHANNELS][LPITCH];  // [buf][ch][j]
    __shared__ __align__(16) unsigned char sink[56 * 16];        // lanes 72..127

    const int tid  = threadIdx.x;
    const int wave = tid >> 5, lane = tid & 31;
    const int g = lane >> 4, ln = lane & 15;
    const int rowBase = blockIdx.x * 64 + wave * 16;
    const int i = rowBase + ln;

    const unsigned zjb = lds_off(&zj[0][0][0]);
    const unsigned sjb = lds_off(&sj[0][0]);
    const unsigned hjb = lds_off(&hjT[0][0][0]);
    const unsigned snb = lds_off(&sink[0]);
    const unsigned long long zbase = (unsigned long long)(uintptr_t)z;
    const unsigned long long qbase = (unsigned long long)(uintptr_t)sq;
    const unsigned long long hbase = (unsigned long long)(uintptr_t)h16T;

    // issue one batch (uniformly 3 async instructions per wave)
    auto issue = [&](int buf, int jc) {
        const unsigned hj = hjb + (unsigned)buf * (CHANNELS * LPITCH * 2);
        #pragma unroll
        for (int it = 0; it < 2; ++it) {
            int c  = tid + it * 128;                 // 256 x 16B chunks
            int ch = c >> 2, sub = c & 3;
            async_copy_b128(hj + ch * (LPITCH * 2) + sub * 16,
                            (unsigned)(ch * (NROWS * 2) + jc * 2 + sub * 16), hbase);
        }
        unsigned l; unsigned long long ga;
        if (tid < 64)      { l = zjb + (unsigned)buf * 1024 + tid * 16;
                             ga = zbase + (unsigned)(jc * 32 + tid * 16); }
        else if (tid < 72) { l = sjb + (unsigned)buf * 128 + (tid - 64) * 16;
                             ga = qbase + (unsigned)(jc * 4 + (tid - 64) * 16); }
        else               { l = snb + (tid - 72) * 16; ga = zbase; }
        async_copy_b128_gv(l, ga);
    };

    float zi[D_LAT];
    #pragma unroll
    for (int d = 0; d < D_LAT; ++d) zi[d] = z[i * D_LAT + d];
    const float sqi = sq[i];

    v8f acc[4] = {};
    float rs = 0.f;

    const float inv2se2 = 1.3888889f;   // 1/(2*0.6^2)
    const float inv2si2 = 0.34722222f;  // 1/(2*1.2^2)
    const int   NCHUNK  = NROWS / 32;   // 256

    issue(0, 0);                        // prologue: batch 0 -> buffer 0
    for (int n = 0; n < NCHUNK; ++n) {
        const int cur = n & 1;
        if (n + 1 < NCHUNK) {
            issue((n + 1) & 1, (n + 1) * 32);
            asm volatile("s_wait_asynccnt 0x3" ::: "memory");  // retire batch n only
        } else {
            asm volatile("s_wait_asynccnt 0x0" ::: "memory");
        }
        __syncthreads();                // batch n visible to all waves

        // K tile 16(i) x 32(j): each lane owns row i = ln
        v16h a;
        #pragma unroll
        for (int e = 0; e < 16; ++e) {
            int k = wk_idx(e, g);
            const float4 za = *(const float4*)&zj[cur][k][0];
            const float4 zb = *(const float4*)&zj[cur][k][4];
            float dot = zi[0] * za.x + zi[1] * za.y + zi[2] * za.z + zi[3] * za.w
                      + zi[4] * zb.x + zi[5] * zb.y + zi[6] * zb.z + zi[7] * zb.w;
            float d2 = fmaxf(sqi + sj[cur][k] - 2.f * dot, 0.f);
            float kv = 0.8f * __expf(-d2 * inv2se2) - __expf(-d2 * inv2si2);
            rs += kv;
            a[e] = (_Float16)kv;
        }
        #pragma unroll
        for (int nt = 0; nt < 4; ++nt) {
            int bc = nt * 16 + ln;
            v8h b01 = *(const v8h*)&hjT[cur][bc][8 * g];
            v8h b23 = *(const v8h*)&hjT[cur][bc][16 + 8 * g];
            acc[nt] = __builtin_amdgcn_wmma_f32_16x16x32_f16(false, a,
                                                             false, cat8(b01, b23),
                                                             (short)0, acc[nt],
                                                             false, false);
        }
        __syncthreads();                // all waves done with buffer cur
    }

    // full rowsum for row ln: combine the two disjoint j-halves (lanes l, l^16)
    rs += __shfl_xor(rs, 16, 32);

    #pragma unroll
    for (int r = 0; r < 8; ++r) {
        // C-fragment row = r + 8*g; its rowsum lives in lane (r + 8*g)
        float rsum = __shfl(rs, (g << 3) + r, 32);
        float scale = 0.06f / (rsum + 1e-6f);   // GAIN / (sum + eps)
        int row = rowBase + r + 8 * g;
        #pragma unroll
        for (int nt = 0; nt < 4; ++nt) {
            int col = nt * 16 + ln;
            h_out[row * CHANNELS + col] =
                h_mid[row * CHANNELS + col] + scale * acc[nt][r];
        }
    }
}

// ---------------------------------------------------------------------------
// Finalize: logits = h @ roW + rob, then pack (logits, z, h) into d_out
// ---------------------------------------------------------------------------
__global__ void __launch_bounds__(256)
finalize_kernel(const float* __restrict__ h, const float* __restrict__ z,
                const float* __restrict__ roW, const float* __restrict__ rob,
                float* __restrict__ out) {
    __shared__ float roWs[CHANNELS * 10];
    __shared__ float robs[10];
    int tid = threadIdx.x;
    for (int i = tid; i < CHANNELS * 10; i += 256) roWs[i] = roW[i];
    if (tid < 10) robs[tid] = rob[tid];
    __syncthreads();

    int idx = blockIdx.x * 256 + tid;
    const int NLOG = NROWS * 10;
    const int NZ   = NROWS * D_LAT;
    const int NH   = NROWS * CHANNELS;
    if (idx < NLOG) {
        int row = idx / 10, c = idx % 10;
        const float4* hp = (const float4*)(h + row * CHANNELS);
        float s = robs[c];
        #pragma unroll
        for (int k4 = 0; k4 < CHANNELS / 4; ++k4) {
            float4 hv = hp[k4];
            s += hv.x * roWs[(k4 * 4 + 0) * 10 + c];
            s += hv.y * roWs[(k4 * 4 + 1) * 10 + c];
            s += hv.z * roWs[(k4 * 4 + 2) * 10 + c];
            s += hv.w * roWs[(k4 * 4 + 3) * 10 + c];
        }
        out[idx] = s;
    } else if (idx < NLOG + NZ) {
        out[idx] = z[idx - NLOG];
    } else if (idx < NLOG + NZ + NH) {
        out[idx] = h[idx - NLOG - NZ];
    }
}

extern "C" void kernel_launch(void* const* d_in, const int* in_sizes, int n_in,
                              void* d_out, int out_size, void* d_ws, size_t ws_size,
                              hipStream_t stream) {
    (void)in_sizes; (void)n_in; (void)out_size; (void)ws_size;
    const float* x       = (const float*)d_in[0];
    const float* W1      = (const float*)d_in[1];
    const float* b1      = (const float*)d_in[2];
    const float* W2      = (const float*)d_in[3];
    const float* b2      = (const float*)d_in[4];
    const float* centers = (const float*)d_in[5];
    const float* mus     = (const float*)d_in[6];
    const float* projW   = (const float*)d_in[7];
    const float* projb   = (const float*)d_in[8];
    const float* roW     = (const float*)d_in[9];
    const float* rob     = (const float*)d_in[10];
    // d_in[11] = T = 5 (reference constant; loop count fixed for graph capture)

    char* ws = (char*)d_ws;
    _Float16* h1   = (_Float16*)(ws + 0);          // 8192*256 f16 = 4 MB
    float*    z    = (float*)(ws + 4194304);       // 8192*8 f32
    float*    sqb  = (float*)(ws + 4456448);       // 8192 f32
    float*    hA   = (float*)(ws + 4489216);       // h_mid  8192*64 f32
    float*    hB   = (float*)(ws + 6586368);       // h_out  8192*64 f32
    _Float16* h16T = (_Float16*)(ws + 8683520);    // transposed [64][8192] f16

    enc1_kernel<<<dim3(NROWS / 16, HID / 64), 128, 0, stream>>>(x, W1, b1, h1);
    enc2_kernel<<<(NROWS * D_LAT) / 256, 256, 0, stream>>>(h1, W2, b2, z);

    for (int t = 0; t < 5; ++t) {
        pm_kernel<<<NROWS / 256, 256, 0, stream>>>(z, centers, mus);
        proj_kernel<<<NROWS / 64, 256, 0, stream>>>(
            z, projW, projb, hB, hA, h16T, sqb, (t == 0) ? 1 : 0);
        lateral_kernel<<<NROWS / 64, 128, 0, stream>>>(z, sqb, h16T, hA, hB);
    }

    int total = NROWS * 10 + NROWS * D_LAT + NROWS * CHANNELS;
    finalize_kernel<<<(total + 255) / 256, 256, 0, stream>>>(hB, z, roW, rob, (float*)d_out);
}
